// LemInFrame_85744727097810
// MI455X (gfx1250) — matrix operand order repaired and checked by reading, not compiled, not run on hardware
//
#include <hip/hip_runtime.h>
#include <hip/hip_bf16.h>

typedef __attribute__((ext_vector_type(16))) __bf16 v16bf;
typedef __attribute__((ext_vector_type(8)))  __bf16 v8bf;
typedef __attribute__((ext_vector_type(8)))  float  v8f;
typedef unsigned short ushort_t;
typedef unsigned int   uint_t;

#define N_EDGES_C   400000
#define N_NODES_C   20000
#define R_MAX_F     5.0f
#define INV_SQRT_AVG 0.22360679775f   // 20^-0.5

// d_out offsets (in floats): latents | node_features | edge_features | cutoff
#define OFF_LAT  0L
#define OFF_NODE 51200000L
#define OFF_EF   61440000L
#define OFF_CUT  266240000L

// ws fragment bases in v16bf (32-byte) units; each frag = 512 bf16 = 32 v16bf
#define FB_L1 0
#define FB_L2 (40*32)
#define FB_L3 (72*32)
#define FB_L4 (104*32)

// LDS row strides (bf16 elements). Chosen so row starts land on distinct banks
// (336B = 84 dw == 20 mod 64 ; 272B = 68 dw == 4 mod 64 -> conflict-free A loads)
// while staying 16B-aligned for ds_load_b128.
#define STRIDE_IN 168
#define STRIDE_H  136

__device__ __forceinline__ ushort_t f2bf(float x) {
    uint_t u = __float_as_uint(x);
    u += 0x7FFFu + ((u >> 16) & 1u);      // round-to-nearest-even
    return (ushort_t)(u >> 16);
}

__device__ __forceinline__ float fast_silu(float x) {
    // x * sigmoid(x), v_rcp_f32 instead of IEEE divide
    return x * __builtin_amdgcn_rcpf(1.0f + __expf(-x));
}

// ---------------- weight prep: f32 -> bf16, pre-swizzled into B-fragment layout ----
// B frag (32x16, bf16): lane l holds col n = l&15 ; element j -> K = (j&7) + 8*(l>>4) + 16*(j>>3)
__global__ void prep_w_kernel(const float* __restrict__ W1, const float* __restrict__ W2,
                              const float* __restrict__ W3, const float* __restrict__ We,
                              ushort_t* __restrict__ ws) {
    int t = blockIdx.x * blockDim.x + threadIdx.x;
    if (t >= 136 * 512) return;
    int f = t >> 9;          // fragment id 0..135
    int e = t & 511;         // element within fragment
    int l = e >> 4;          // lane 0..31
    int j = e & 15;          // element within lane
    const float* src; int ntile, kstep, K;
    if (f < 40) { src = W1; ntile = f / 5; kstep = f % 5; K = 136; }
    else {
        int f2 = f - 40;
        int layer = f2 >> 5;                // 0=W2, 1=W3, 2=W_env
        int r = f2 & 31;
        ntile = r >> 2; kstep = r & 3; K = 128;
        src = (layer == 0) ? W2 : (layer == 1) ? W3 : We;
    }
    int k = kstep * 32 + (j & 7) + 8 * (l >> 4) + 16 * (j >> 3);
    int n = ntile * 16 + (l & 15);
    float v = (k < K) ? src[k * 128 + n] : 0.0f;
    ws[f * 512 + l * 16 + j] = f2bf(v);
}

// A frag (16x32, bf16) gather per ISA layout: lane<16 row=lane, K {kb..kb+7, kb+16..kb+23};
// lanes 16..31 same rows, +8 on each K run.  Two aligned 16B LDS loads.
__device__ __forceinline__ v16bf load_a(const ushort_t* buf, int stride, int kbase,
                                        int lr, int hi) {
    const v8bf* p0 = (const v8bf*)(buf + lr * stride + kbase + 8 * hi);
    const v8bf* p1 = (const v8bf*)(buf + lr * stride + kbase + 16 + 8 * hi);
    v8bf a = *p0, b = *p1;
    return __builtin_shufflevector(a, b, 0,1,2,3,4,5,6,7,8,9,10,11,12,13,14,15);
}

__global__ __launch_bounds__(128)
void edge_mlp_env_kernel(const int*   __restrict__ edge_index,
                         const float* __restrict__ edge_sh,
                         const float* __restrict__ edge_length,
                         const float* __restrict__ edge_one_hot,
                         const float* __restrict__ bessel_w,
                         const ushort_t* __restrict__ ws,
                         float* __restrict__ out) {
    __shared__ __align__(16) ushort_t s_in[4][16 * STRIDE_IN]; // layer1 A / layer2 dest
    __shared__ __align__(16) ushort_t s_h [4][16 * STRIDE_H];  // layer1 dest / latents bf16
    __shared__ __align__(16) float    s_w [4][16 * 32];        // weights_e per l-group pair
    __shared__ __align__(16) float    s_sh[4][16 * 16];        // sh tile
    __shared__ float s_cut[4][16];
    __shared__ int   s_cen[4][16];

    const int w    = threadIdx.x >> 5;
    const int lane = threadIdx.x & 31;
    const int hi   = lane >> 4;
    const int lr   = lane & 15;
    const int tile = blockIdx.x * 4 + w;
    const long e0  = (long)tile * 16;

    // ---------------- stage: cutoff, center, bessel, sh, one_hot -------------------
    if (lane < 16) {
        long e = e0 + lane;
        float r = edge_length[e];
        float x = r * (1.0f / R_MAX_F);
        float x2 = x * x;
        float x6 = x2 * x2 * x2;
        float c = 1.0f - 28.0f * x6 + 48.0f * x6 * x - 21.0f * x6 * x2;
        c = (x < 1.0f) ? c : 0.0f;
        s_cut[w][lane] = c;
        s_cen[w][lane] = edge_index[e];   // edge_index row 0 = centers
        out[OFF_CUT + e] = c;
        float inv_r = __builtin_amdgcn_rcpf(r);
        #pragma unroll
        for (int j = 0; j < 8; ++j) {
            float s  = __sinf(bessel_w[j] * x);
            float rb = (2.0f / R_MAX_F) * s * inv_r;
            s_in[w][lane * STRIDE_IN + 128 + j] = f2bf(rb);
        }
        #pragma unroll
        for (int t = 0; t < 12; ++t)      // zero pad K = 136..159
            *(uint_t*)&s_in[w][lane * STRIDE_IN + 136 + 2 * t] = 0u;
    }
    {   // sh tile: 16 edges x 16 f32, contiguous -> 64 float4
        const float4* shp = (const float4*)(edge_sh + e0 * 16);
        #pragma unroll
        for (int it = 0; it < 2; ++it) {
            int idx = lane + 32 * it;
            float4 v = shp[idx];
            *(float4*)&s_sh[w][idx * 4] = v;
        }
    }
    {   // one_hot: 16 x 128 f32 -> bf16 into s_in cols 0..127
        const float4* op = (const float4*)(edge_one_hot + e0 * 128);
        #pragma unroll
        for (int it = 0; it < 16; ++it) {
            int idx = lane + 32 * it;        // 0..511
            int row = idx >> 5, c4 = idx & 31;
            float4 v = op[idx];
            uint2 pk;
            pk.x = (uint_t)f2bf(v.x) | ((uint_t)f2bf(v.y) << 16);
            pk.y = (uint_t)f2bf(v.z) | ((uint_t)f2bf(v.w) << 16);
            *(uint2*)&s_in[w][row * STRIDE_IN + c4 * 4] = pk;
        }
    }
    __syncthreads();

    const v16bf* wsv = (const v16bf*)ws;

    // ---------------- layer 1: [16x160] x [160x128] -> silu -> s_h -----------------
    for (int nt = 0; nt < 8; ++nt) {
        v8f acc = {};
        #pragma unroll
        for (int ks = 0; ks < 5; ++ks) {
            v16bf a = load_a(&s_in[w][0], STRIDE_IN, ks * 32, lr, hi);
            v16bf b = wsv[FB_L1 + (nt * 5 + ks) * 32 + lane];
            acc = __builtin_amdgcn_wmma_f32_16x16x32_bf16(false, a, false, b,
                                                          (short)0, acc, false, false);
        }
        #pragma unroll
        for (int i = 0; i < 8; ++i) {
            int m = i + 8 * hi;
            s_h[w][m * STRIDE_H + nt * 16 + lr] = f2bf(fast_silu(acc[i]));
        }
    }

    // ---------------- layer 2: s_h x W2 -> silu -> s_in ----------------------------
    for (int nt = 0; nt < 8; ++nt) {
        v8f acc = {};
        #pragma unroll
        for (int ks = 0; ks < 4; ++ks) {
            v16bf a = load_a(&s_h[w][0], STRIDE_H, ks * 32, lr, hi);
            v16bf b = wsv[FB_L2 + (nt * 4 + ks) * 32 + lane];
            acc = __builtin_amdgcn_wmma_f32_16x16x32_bf16(false, a, false, b,
                                                          (short)0, acc, false, false);
        }
        #pragma unroll
        for (int i = 0; i < 8; ++i) {
            int m = i + 8 * hi;
            s_in[w][m * STRIDE_IN + nt * 16 + lr] = f2bf(fast_silu(acc[i]));
        }
    }

    // ---------------- layer 3: s_in x W3 -> cutoff mask -> latents (global + s_h) --
    for (int nt = 0; nt < 8; ++nt) {
        v8f acc = {};
        #pragma unroll
        for (int ks = 0; ks < 4; ++ks) {
            v16bf a = load_a(&s_in[w][0], STRIDE_IN, ks * 32, lr, hi);
            v16bf b = wsv[FB_L3 + (nt * 4 + ks) * 32 + lane];
            acc = __builtin_amdgcn_wmma_f32_16x16x32_bf16(false, a, false, b,
                                                          (short)0, acc, false, false);
        }
        #pragma unroll
        for (int i = 0; i < 8; ++i) {
            int m = i + 8 * hi;
            float c = s_cut[w][m];
            float lat = (c > 0.0f) ? c * acc[i] : 0.0f;
            out[OFF_LAT + (e0 + m) * 128 + nt * 16 + lr] = lat;
            s_h[w][m * STRIDE_H + nt * 16 + lr] = f2bf(lat);
        }
    }

    // ---------------- layer 4 (W_env) interleaved with env-weighting ---------------
    const int dar[4]   = {1, 3, 5, 7};
    const int shoff[4] = {0, 1, 4, 9};
    const int ooff[4]  = {0, 32, 128, 288};
    #pragma unroll
    for (int g = 0; g < 4; ++g) {
        #pragma unroll
        for (int sub = 0; sub < 2; ++sub) {
            int nt = 2 * g + sub;
            v8f acc = {};
            #pragma unroll
            for (int ks = 0; ks < 4; ++ks) {
                v16bf a = load_a(&s_h[w][0], STRIDE_H, ks * 32, lr, hi);
                v16bf b = wsv[FB_L4 + (nt * 4 + ks) * 32 + lane];
                acc = __builtin_amdgcn_wmma_f32_16x16x32_bf16(false, a, false, b,
                                                              (short)0, acc, false, false);
            }
            #pragma unroll
            for (int i = 0; i < 8; ++i) {
                int m = i + 8 * hi;
                s_w[w][m * 32 + sub * 16 + lr] = acc[i];
            }
        }
        // env: edge_features[e, ooff + mul*d + k] = w[e,mul] * sh[e, shoff + k]
        int d = dar[g];
        int W = 32 * d;
        int iters = 16 * d;                 // 16 edges * W outputs / 32 lanes
        for (int it = 0; it < iters; ++it) {
            int flat = lane + 32 * it;
            int er   = flat / W;
            int rem  = flat - er * W;
            int mul  = rem / d;
            int k    = rem - mul * d;
            float v = s_w[w][er * 32 + mul] * s_sh[w][er * 16 + shoff[g] + k];
            out[OFF_EF + (e0 + er) * 512 + ooff[g] + rem] = v;
            atomicAdd(&out[OFF_NODE + (long)s_cen[w][er] * 512 + ooff[g] + rem],
                      v * INV_SQRT_AVG);
        }
    }
}

extern "C" void kernel_launch(void* const* d_in, const int* in_sizes, int n_in,
                              void* d_out, int out_size, void* d_ws, size_t ws_size,
                              hipStream_t stream) {
    const int*   edge_index   = (const int*)d_in[0];
    // d_in[1] atom_type, d_in[2] bond_type: unused by reference
    const float* edge_sh      = (const float*)d_in[3];
    const float* edge_length  = (const float*)d_in[4];
    const float* edge_one_hot = (const float*)d_in[5];
    const float* bessel_w     = (const float*)d_in[6];
    const float* W1           = (const float*)d_in[7];
    const float* W2           = (const float*)d_in[8];
    const float* W3           = (const float*)d_in[9];
    const float* We           = (const float*)d_in[10];
    float* out   = (float*)d_out;
    ushort_t* ws = (ushort_t*)d_ws;

    // node_features is accumulated with atomics -> zero it every launch (capturable)
    hipMemsetAsync(out + OFF_NODE, 0, (size_t)N_NODES_C * 512 * sizeof(float), stream);

    prep_w_kernel<<<(136 * 512) / 256, 256, 0, stream>>>(W1, W2, W3, We, ws);

    // 400000 edges = 25000 tiles of 16; 4 waves (tiles) per 128-thread block
    edge_mlp_env_kernel<<<25000 / 4, 128, 0, stream>>>(
        edge_index, edge_sh, edge_length, edge_one_hot, bessel_w, ws, out);
}
